// GroupedQueryAttention_41257455846150
// MI455X (gfx1250) — matrix operand order repaired and checked by reading, not compiled, not run on hardware
//
#include <hip/hip_runtime.h>

// ---------------- problem constants ----------------
static constexpr int Bb  = 2;
static constexpr int Ss  = 2048;
static constexpr int Hh  = 2048;
static constexpr int NQ  = 32;
static constexpr int NKV = 8;
static constexpr int HD  = 64;          // head dim
static constexpr int Gg  = NQ / NKV;    // 4
static constexpr int Mm  = Bb * Ss;     // 4096 rows
static constexpr int KV  = NKV * HD;    // 512

// ---------------- vector types ----------------
typedef __attribute__((ext_vector_type(16))) _Float16 v16h;
typedef __attribute__((ext_vector_type(8)))  _Float16 v8h;
typedef __attribute__((ext_vector_type(8)))  float    v8f;

// Load a 16-f16 WMMA fragment from two contiguous 8-f16 (16B) chunks.
__device__ __forceinline__ v16h ld_frag16(const _Float16* p0, const _Float16* p1) {
  union { v16h v; v8h h[2]; } u;
  u.h[0] = *(const v8h*)p0;
  u.h[1] = *(const v8h*)p1;
  return u.v;
}

__device__ __forceinline__ v8f wmma_f16(v16h a, v16h b, v8f c) {
  return __builtin_amdgcn_wmma_f32_16x16x32_f16(false, a, false, b, (short)0, c,
                                                false, false);
}

// CDNA5 async copy: 16 bytes global -> LDS, tracked by ASYNCcnt.
// LDS operand = low 32 bits of the generic shared address (wave-relative LDS offset).
__device__ __forceinline__ void async_b128_to_lds(const _Float16* g, _Float16* l) {
  uint32_t loff = (uint32_t)(uintptr_t)l;
  uint64_t gaddr = (uint64_t)(uintptr_t)g;
  asm volatile("global_load_async_to_lds_b128 %0, %1, off"
               :: "v"(loff), "v"(gaddr) : "memory");
}

__device__ __forceinline__ void wait_async0() {
  asm volatile("s_wait_asynccnt 0" ::: "memory");
}

// ---------------- prep kernels ----------------
__global__ void k_f32_to_f16(const float* __restrict__ in, _Float16* __restrict__ out, int n) {
  int i = blockIdx.x * blockDim.x + threadIdx.x;
  if (i < n) out[i] = (_Float16)in[i];
}

// W [K x N] fp32 -> WT [N x K] f16
__global__ void k_transpose_w(const float* __restrict__ in, _Float16* __restrict__ out,
                              int Kd, int Nd) {
  int i = blockIdx.x * blockDim.x + threadIdx.x;
  if (i >= Kd * Nd) return;
  int k = i % Kd, n = i / Kd;
  out[(size_t)n * Kd + k] = (_Float16)in[(size_t)k * Nd + n];
}

// V f16 [B][S][NKV][HD] -> VT f16 [B][NKV][HD][S]
__global__ void k_transpose_v(const _Float16* __restrict__ v, _Float16* __restrict__ vt) {
  int i = blockIdx.x * blockDim.x + threadIdx.x;
  int total = Bb * NKV * HD * Ss;
  if (i >= total) return;
  int s = i % Ss;
  int d = (i / Ss) % HD;
  int h = (i / (Ss * HD)) % NKV;
  int b = i / (Ss * HD * NKV);
  vt[i] = v[((size_t)(b * Ss + s)) * KV + h * HD + d];
}

// ---------------- WMMA GEMM: C[MxN] = A[MxK](f16) @ BT([N][K] f16) + bias ----------------
// 8 waves / block; block tile = 128(M) x 64(N); grid = (M/128, N/64).
// B tile (shared by all 8 waves) is double-buffered in LDS via async copies;
// A fragments are double-buffered in registers. All 4 B fragments are loaded
// into distinct registers before the WMMA burst so the ds_loads clause together.
__global__ __launch_bounds__(256)
void k_gemm_wmma(const _Float16* __restrict__ A, const _Float16* __restrict__ BT,
                 const float* __restrict__ bias, _Float16* __restrict__ Cf16,
                 float* __restrict__ Cf32, int Md, int Nd, int Kd, int f32out) {
  const int tid  = threadIdx.x;
  const int wave = tid >> 5;
  const int lane = tid & 31;
  const int l15  = lane & 15;
  const int lh   = lane >> 4;
  const int row0 = (blockIdx.x * 8 + wave) * 16;
  const int n0   = blockIdx.y * 64;

  // [buf][col][k] : 64 cols x 32 k of f16 = 4KB per buffer
  __shared__ __align__(16) _Float16 Btile[2][64 * 32];

  // one async b128 per thread fills a whole 4KB buffer
  const int sc = tid >> 2;        // column 0..63
  const int sk = (tid & 3) * 8;   // k-chunk 0,8,16,24
  const _Float16* bsrc = BT + (size_t)(n0 + sc) * Kd + sk;

  v8f acc[4] = {};
  const _Float16* arow = A + (size_t)(row0 + l15) * Kd;
  const int akl = lh * 16;

  // prologue: stage k0=0, preload first A fragment
  async_b128_to_lds(bsrc, &Btile[0][sc * 32 + sk]);
  v16h afc = ld_frag16(arow + akl, arow + akl + 8);
  wait_async0();
  __syncthreads();

  for (int k0 = 0; k0 < Kd; k0 += 32) {
    const int cur  = (k0 >> 5) & 1;
    const int nxt  = cur ^ 1;
    const bool more = (k0 + 32) < Kd;

    // stage next B tile (async, no wait yet) and next A fragment
    if (more) {
      async_b128_to_lds(bsrc + k0 + 32, &Btile[nxt][sc * 32 + sk]);
      __builtin_prefetch(arow + k0 + 288, 0, 1);  // global_prefetch_b8
    }
    v16h afn;
    if (more) afn = ld_frag16(arow + k0 + 32 + akl, arow + k0 + 32 + akl + 8);

    // load all 4 B fragments (8 x ds_load_b128 clause), then WMMA burst
    v16h bf[4];
#pragma unroll
    for (int t = 0; t < 4; ++t) {
      const _Float16* bp = &Btile[cur][(t * 16 + l15) * 32 + lh * 16];
      bf[t] = ld_frag16(bp, bp + 8);
    }
#pragma unroll
    for (int t = 0; t < 4; ++t) acc[t] = wmma_f16(afc, bf[t], acc[t]);

    if (more) {
      wait_async0();
      __syncthreads();
      afc = afn;
    }
  }

#pragma unroll
  for (int t = 0; t < 4; ++t) {
    const int col = n0 + t * 16 + l15;
    const float bv = bias[col];
#pragma unroll
    for (int r = 0; r < 8; ++r) {
      const int row = row0 + r + 8 * lh;   // C layout: M = r + 8*(lane/16), N = lane%16
      const float val = acc[t][r] + bv;
      if (f32out) Cf32[(size_t)row * Nd + col] = val;
      else        Cf16[(size_t)row * Nd + col] = (_Float16)val;
    }
  }
}

// ---------------- grouped-query flash attention ----------------
// one wave (32 threads) per (b, q-head, 16-query tile); grid = B*NQ*(S/16)
__global__ __launch_bounds__(32)
void k_gqa_attn(const _Float16* __restrict__ Q,   // [M][H]  (head-major cols)
                const _Float16* __restrict__ Kf,  // [M][KV]
                const _Float16* __restrict__ VT,  // [B][NKV][HD][S]
                _Float16* __restrict__ AO) {      // [M][H]
  const int QT  = Ss / 16;                  // 128
  const int qt  = blockIdx.x % QT;
  const int hq  = (blockIdx.x / QT) % NQ;
  const int b   = blockIdx.x / (QT * NQ);
  const int kvh = hq / Gg;

  const int lane = threadIdx.x;
  const int l15  = lane & 15;
  const int lh   = lane >> 4;
  const int akl  = lh * 16;

  __shared__ __align__(16) _Float16 pT[16 * 32];

  // Q A-fragments for HD reduction (two 16x32 k-slabs)
  const _Float16* qbase = Q + (size_t)(b * Ss + qt * 16 + l15) * Hh + hq * HD;
  const v16h aq0 = ld_frag16(qbase + akl,      qbase + akl + 8);
  const v16h aq1 = ld_frag16(qbase + 32 + akl, qbase + 32 + akl + 8);

  float mstate[8], lstate[8];
#pragma unroll
  for (int r = 0; r < 8; ++r) { mstate[r] = -1e30f; lstate[r] = 0.f; }
  v8f acc[4] = {};

  const _Float16* kbase  = Kf + (size_t)b * Ss * KV + kvh * HD;
  const _Float16* vtbase = VT + ((size_t)(b * NKV + kvh) * HD) * Ss;
  const float scale = 0.125f;   // 1/sqrt(64)

  for (int j0 = 0; j0 < Ss; j0 += 32) {
    // ---- issue V fragment loads first so they overlap scores+softmax ----
    v16h bv[4];
#pragma unroll
    for (int t = 0; t < 4; ++t) {
      const _Float16* vp = vtbase + (size_t)(t * 16 + l15) * Ss + j0 + lh * 16;
      bv[t] = ld_frag16(vp, vp + 8);
    }

    // ---- scores: 16 queries x 32 keys, HD=64 reduction ----
    // load all 4 K fragments into distinct regs, then WMMA burst
    v8f c0 = {}, c1 = {};
    {
      const _Float16* kp0 = kbase + (size_t)(j0 + l15) * KV + lh * 16;
      const _Float16* kp1 = kbase + (size_t)(j0 + 16 + l15) * KV + lh * 16;
      v16h bk00 = ld_frag16(kp0,      kp0 + 8);
      v16h bk01 = ld_frag16(kp0 + 32, kp0 + 40);
      v16h bk10 = ld_frag16(kp1,      kp1 + 8);
      v16h bk11 = ld_frag16(kp1 + 32, kp1 + 40);
      c0 = wmma_f16(aq0, bk00, c0);
      c0 = wmma_f16(aq1, bk01, c0);
      c1 = wmma_f16(aq0, bk10, c1);
      c1 = wmma_f16(aq1, bk11, c1);
    }

    // ---- online softmax (row lives across the 16 lanes of one half) ----
    float alpha[8];
#pragma unroll
    for (int r = 0; r < 8; ++r) {
      float s0 = c0[r] * scale, s1 = c1[r] * scale;
      float mx = fmaxf(s0, s1);
#pragma unroll
      for (int off = 8; off; off >>= 1) mx = fmaxf(mx, __shfl_xor(mx, off, 16));
      const float mn = fmaxf(mstate[r], mx);
      alpha[r]  = __expf(mstate[r] - mn);
      mstate[r] = mn;
      const float p0 = __expf(s0 - mn), p1 = __expf(s1 - mn);
      float rs = p0 + p1;
#pragma unroll
      for (int off = 8; off; off >>= 1) rs += __shfl_xor(rs, off, 16);
      lstate[r] = lstate[r] * alpha[r] + rs;
      pT[(r + 8 * lh) * 32 + l15]      = (_Float16)p0;
      pT[(r + 8 * lh) * 32 + 16 + l15] = (_Float16)p1;
    }
    __syncthreads();
    const v16h pA = ld_frag16(&pT[l15 * 32 + akl], &pT[l15 * 32 + akl + 8]);
    __syncthreads();

    // ---- rescale accumulator, then acc += P @ V ----
#pragma unroll
    for (int t = 0; t < 4; ++t)
#pragma unroll
      for (int r = 0; r < 8; ++r) acc[t][r] *= alpha[r];
#pragma unroll
    for (int t = 0; t < 4; ++t) acc[t] = wmma_f16(pA, bv[t], acc[t]);
  }

  // ---- normalize and write out ----
#pragma unroll
  for (int t = 0; t < 4; ++t) {
    const int hd = t * 16 + l15;
#pragma unroll
    for (int r = 0; r < 8; ++r) {
      const int q = qt * 16 + r + 8 * lh;
      AO[(size_t)(b * Ss + q) * Hh + hq * HD + hd] =
          (_Float16)(acc[t][r] / lstate[r]);
    }
  }
}

// ---------------- host launcher ----------------
extern "C" void kernel_launch(void* const* d_in, const int* in_sizes, int n_in,
                              void* d_out, int out_size, void* d_ws, size_t ws_size,
                              hipStream_t stream) {
  const float* x  = (const float*)d_in[0];
  const float* Wq = (const float*)d_in[1];
  const float* bq = (const float*)d_in[2];
  const float* Wk = (const float*)d_in[3];
  const float* bk = (const float*)d_in[4];
  const float* Wv = (const float*)d_in[5];
  const float* bv = (const float*)d_in[6];
  const float* Wo = (const float*)d_in[7];
  const float* bo = (const float*)d_in[8];

  char* ws = (char*)d_ws;
  // f16 scratch layout (bytes)
  constexpr size_t nXH  = (size_t)Mm * Hh;   // x in f16
  constexpr size_t nWQT = (size_t)Hh * Hh;
  constexpr size_t nWKT = (size_t)KV * Hh;
  constexpr size_t nWOT = (size_t)Hh * Hh;
  constexpr size_t nQF  = (size_t)Mm * Hh;
  constexpr size_t nKF  = (size_t)Mm * KV;
  constexpr size_t nAO  = (size_t)Mm * Hh;

  size_t off = 0;
  _Float16* XH  = (_Float16*)(ws + off); off += nXH  * 2;
  _Float16* WQT = (_Float16*)(ws + off); off += nWQT * 2;
  _Float16* WKT = (_Float16*)(ws + off); off += nWKT * 2;
  _Float16* WVT = (_Float16*)(ws + off); off += nWKT * 2;
  _Float16* WOT = (_Float16*)(ws + off); off += nWOT * 2;
  _Float16* QF  = (_Float16*)(ws + off); off += nQF  * 2;
  _Float16* KF  = (_Float16*)(ws + off); off += nKF  * 2;
  _Float16* VF  = (_Float16*)(ws + off); off += nKF  * 2;
  _Float16* VTT = (_Float16*)(ws + off); off += nKF  * 2;
  _Float16* AO  = (_Float16*)(ws + off); off += nAO  * 2;

  // 1) convert activations and weights
  k_f32_to_f16<<<(int)((nXH + 255) / 256), 256, 0, stream>>>(x, XH, (int)nXH);
  k_transpose_w<<<(int)((nWQT + 255) / 256), 256, 0, stream>>>(Wq, WQT, Hh, Hh);
  k_transpose_w<<<(int)((nWKT + 255) / 256), 256, 0, stream>>>(Wk, WKT, Hh, KV);
  k_transpose_w<<<(int)((nWKT + 255) / 256), 256, 0, stream>>>(Wv, WVT, Hh, KV);
  k_transpose_w<<<(int)((nWOT + 255) / 256), 256, 0, stream>>>(Wo, WOT, Hh, Hh);

  // 2) QKV projections (WMMA GEMMs, f16 outputs)
  dim3 gq(Mm / 128, Hh / 64);
  k_gemm_wmma<<<gq, 256, 0, stream>>>(XH, WQT, bq, QF, nullptr, Mm, Hh, Hh, 0);
  dim3 gk(Mm / 128, KV / 64);
  k_gemm_wmma<<<gk, 256, 0, stream>>>(XH, WKT, bk, KF, nullptr, Mm, KV, Hh, 0);
  k_gemm_wmma<<<gk, 256, 0, stream>>>(XH, WVT, bv, VF, nullptr, Mm, KV, Hh, 0);

  // 3) V transpose for contiguous PV B-fragments
  k_transpose_v<<<(int)((nKF + 255) / 256), 256, 0, stream>>>(VF, VTT);

  // 4) grouped-query flash attention
  k_gqa_attn<<<Bb * NQ * (Ss / 16), 32, 0, stream>>>(QF, KF, VTT, AO);

  // 5) output projection (fp32 out)
  k_gemm_wmma<<<gq, 256, 0, stream>>>(AO, WOT, bo, nullptr, (float*)d_out,
                                      Mm, Hh, Hh, 1);
}
